// GatedPooling_54683523612692
// MI455X (gfx1250) — matrix operand order repaired and checked by reading, not compiled, not run on hardware
//
#include <hip/hip_runtime.h>

// ---------------------------------------------------------------------------
// GatedPooling for MI455X (gfx1250, wave32, WMMA)
//   pooled[b,e] = sum_t  w[b,t] * x[b,t,e] * sigmoid( sum_d x[b,t,d]*W[e,d] + bias[e] )
//   w = sparsemax (entmax alpha=2) over T via 50 bisection iterations.
// GEMM runs on v_wmma_f32_16x16x32_bf16 (f32 accumulate), K staged 64-wide in LDS.
// ---------------------------------------------------------------------------

typedef __attribute__((ext_vector_type(16))) __bf16 v16bf;
typedef __attribute__((ext_vector_type(8)))  float  v8f;

#define TILE_M 128
#define TILE_N 128
#define TILE_K 64
#define LDS_STRIDE 72      // 64 + 8 pad bf16 elems; 144B rows keep 16B alignment

union FragBF {
    v16bf v;
    uint4 q[2];
};

#if defined(__AMDGCN__) && __has_builtin(__builtin_amdgcn_cvt_pk_bf16_f32)
typedef __attribute__((ext_vector_type(2))) __bf16 v2bf;
__device__ __forceinline__ unsigned pack_bf16(float a, float b) {
    union { v2bf v; unsigned u; } cv;
    cv.v = __builtin_amdgcn_cvt_pk_bf16_f32(a, b);   // v_cvt_pk_bf16_f32
    return cv.u;
}
#else
__device__ __forceinline__ unsigned pack_bf16(float a, float b) {
    // round-half-up f32->bf16 pair pack: 2x v_add_nc_u32 + 1x v_perm_b32.
    // D[15:0] = hi16(a+0x8000), D[31:16] = hi16(b+0x8000).
    unsigned ap = __float_as_uint(a) + 0x8000u;
    unsigned bp = __float_as_uint(b) + 0x8000u;
    return __builtin_amdgcn_perm(bp, ap, 0x07060302u);
}
#endif

__device__ __forceinline__ float fast_sigmoid(float v) {
    // v_exp_f32 + v_rcp_f32 (avoid IEEE divide expansion)
    return __builtin_amdgcn_rcpf(1.0f + __expf(-v));
}

// ---------------------------------------------------------------------------
// Kernel 1: sparsemax (entmax alpha=2) by bisection, one block per batch row.
// ---------------------------------------------------------------------------
#define ENT_THREADS 256
#define ENT_MAXCH   8

__device__ __forceinline__ float waveReduceSum(float v) {
#pragma unroll
    for (int o = 16; o > 0; o >>= 1) v += __shfl_xor(v, o, 32);
    return v;
}
__device__ __forceinline__ float waveReduceMax(float v) {
#pragma unroll
    for (int o = 16; o > 0; o >>= 1) v = fmaxf(v, __shfl_xor(v, o, 32));
    return v;
}
__device__ float blockReduceSum(float v, float* lds) {
    const int tid = threadIdx.x;
    v = waveReduceSum(v);
    __syncthreads();
    if ((tid & 31) == 0) lds[tid >> 5] = v;
    __syncthreads();
    if (tid == 0) {
        float s = lds[0];
        for (int i = 1; i < ENT_THREADS / 32; ++i) s += lds[i];
        lds[ENT_THREADS / 32] = s;
    }
    __syncthreads();
    return lds[ENT_THREADS / 32];
}
__device__ float blockReduceMax(float v, float* lds) {
    const int tid = threadIdx.x;
    v = waveReduceMax(v);
    __syncthreads();
    if ((tid & 31) == 0) lds[tid >> 5] = v;
    __syncthreads();
    if (tid == 0) {
        float s = lds[0];
        for (int i = 1; i < ENT_THREADS / 32; ++i) s = fmaxf(s, lds[i]);
        lds[ENT_THREADS / 32] = s;
    }
    __syncthreads();
    return lds[ENT_THREADS / 32];
}

__global__ __launch_bounds__(ENT_THREADS)
void gp_entmax_kernel(const float* __restrict__ scores,
                      float* __restrict__ attnw, int T) {
    __shared__ float lds[ENT_THREADS / 32 + 1];
    const int b = blockIdx.x;
    const int tid = threadIdx.x;

    float Xv[ENT_MAXCH];
    int n = 0;
    for (int t = tid; t < T; t += ENT_THREADS) Xv[n++] = scores[(size_t)b * T + t];

    float m = -3.402823466e38f;
    for (int i = 0; i < n; ++i) m = fmaxf(m, Xv[i]);
    const float mx = blockReduceMax(m, lds);

    float tau_lo = mx - 1.0f;                   // max - 1^(alpha-1)
    const float tau_hi = mx - 1.0f / (float)T;  // max - (1/d)^(alpha-1)

    float loc = 0.f;
    for (int i = 0; i < n; ++i) loc += fmaxf(Xv[i] - tau_lo, 0.f);
    const float f_lo = blockReduceSum(loc, lds) - 1.0f;

    float dm = tau_hi - tau_lo;
    float tau_m = tau_lo, f_m = 0.f;
    for (int it = 0; it < 50; ++it) {
        dm *= 0.5f;
        tau_m = tau_lo + dm;
        loc = 0.f;
        for (int i = 0; i < n; ++i) loc += fmaxf(Xv[i] - tau_m, 0.f);
        f_m = blockReduceSum(loc, lds) - 1.0f;
        if (f_m * f_lo >= 0.f) tau_lo = tau_m;   // uniform across block
    }
    const float inv = 1.0f / (f_m + 1.0f);       // ensure_sum_one
    n = 0;
    for (int t = tid; t < T; t += ENT_THREADS)
        attnw[(size_t)b * T + t] = fmaxf(Xv[n++] - tau_m, 0.f) * inv;
}

// ---------------------------------------------------------------------------
// Kernel 2: fused GEMM(x @ W^T) + sigmoid gate + weighted pooling partials.
// Block = 256 threads (8 waves), tile 128(M=t) x 128(N=e), K staged 64-wide.
// Wave (wm 0..3, wn 0..1) owns a 32x64 sub-tile = 2x4 WMMA 16x16 tiles,
// issuing 16 v_wmma_f32_16x16x32_bf16 per LDS stage (2 k-steps).
// ---------------------------------------------------------------------------
__global__ __launch_bounds__(256)
void gp_gemm_pool_kernel(const float* __restrict__ x,
                         const float* __restrict__ gw,
                         const float* __restrict__ gb,
                         const float* __restrict__ attnw,
                         float* __restrict__ partial,
                         int T, int D, int MB) {
    const int b    = blockIdx.z;
    const int t0   = blockIdx.y * TILE_M;
    const int e0   = blockIdx.x * TILE_N;
    const int tid  = threadIdx.x;
    const int lane = tid & 31;
    const int wave = tid >> 5;
    const int wm   = wave >> 1;     // M group (32 rows of t)
    const int wn   = wave & 1;      // N group (64 cols of e)
    const int laneN = lane & 15;
    const int hi    = lane >> 4;    // lane half (ISA WMMA layouts)

    __shared__ __align__(16) unsigned short As[TILE_M * LDS_STRIDE]; // x tile, bf16
    __shared__ __align__(16) unsigned short Bs[TILE_N * LDS_STRIDE]; // W tile, bf16
    __shared__ float red[4][TILE_N];

    const float* xA = x  + ((size_t)b * T + t0) * (size_t)D;
    const float* wB = gw + (size_t)e0 * (size_t)D;

    const v8f zero = {0.f, 0.f, 0.f, 0.f, 0.f, 0.f, 0.f, 0.f};
    v8f acc[2][4];
#pragma unroll
    for (int mt = 0; mt < 2; ++mt)
#pragma unroll
        for (int nt = 0; nt < 4; ++nt) acc[mt][nt] = zero;

    for (int k0 = 0; k0 < D; k0 += TILE_K) {
        // Stage A and B panels (f32 -> bf16). 128x64 each: 2048 float4 per
        // panel, 8 float4 per thread per panel.
#pragma unroll
        for (int j = 0; j < 8; ++j) {
            const int f   = tid + 256 * j;     // float4 index within panel
            const int row = f >> 4;            // 16 float4 per 64-wide row
            const int col = (f & 15) << 2;
            float4 a = *reinterpret_cast<const float4*>(xA + (size_t)row * D + k0 + col);
            uint2 pa;
            pa.x = pack_bf16(a.x, a.y);
            pa.y = pack_bf16(a.z, a.w);
            *reinterpret_cast<uint2*>(&As[row * LDS_STRIDE + col]) = pa;
            float4 w4 = *reinterpret_cast<const float4*>(wB + (size_t)row * D + k0 + col);
            uint2 pb;
            pb.x = pack_bf16(w4.x, w4.y);
            pb.y = pack_bf16(w4.z, w4.w);
            *reinterpret_cast<uint2*>(&Bs[row * LDS_STRIDE + col]) = pb;
        }
        __syncthreads();

        // Fragments per ISA VGPR layouts (wave32):
        //  A 16x32 bf16: lane -> row M=lane%16; K chunks {kb..kb+7, kb+16..kb+23}, kb=hi*8
        //  B 32x16 bf16: lane -> col N=lane%16; K chunks {kb..kb+7, kb+16..kb+23}, kb=hi*8
#pragma unroll
        for (int ks = 0; ks < 2; ++ks) {       // two k-steps per stage
            const int kbase = ks * 32;
            FragBF afr[2], bfr[4];
#pragma unroll
            for (int mt = 0; mt < 2; ++mt) {
                const int arow = wm * 32 + mt * 16 + laneN;
                const int ka   = kbase + hi * 8;
                afr[mt].q[0] = *reinterpret_cast<const uint4*>(&As[arow * LDS_STRIDE + ka]);
                afr[mt].q[1] = *reinterpret_cast<const uint4*>(&As[arow * LDS_STRIDE + ka + 16]);
            }
#pragma unroll
            for (int nt = 0; nt < 4; ++nt) {
                const int brow = wn * 64 + nt * 16 + laneN;
                const int kb   = kbase + hi * 8;
                bfr[nt].q[0] = *reinterpret_cast<const uint4*>(&Bs[brow * LDS_STRIDE + kb]);
                bfr[nt].q[1] = *reinterpret_cast<const uint4*>(&Bs[brow * LDS_STRIDE + kb + 16]);
            }
#pragma unroll
            for (int mt = 0; mt < 2; ++mt)
#pragma unroll
                for (int nt = 0; nt < 4; ++nt)
                    acc[mt][nt] = __builtin_amdgcn_wmma_f32_16x16x32_bf16(
                        false, afr[mt].v, false, bfr[nt].v,
                        (short)0, acc[mt][nt], false, false);
        }
        __syncthreads();
    }

    // Epilogue: gate = sigmoid(acc + bias); contribution = w[t] * x[t,e] * gate.
    // C/D layout: VGPR r -> M = r + hi*8, N = lane%16.
    float biasv[4];
#pragma unroll
    for (int nt = 0; nt < 4; ++nt)
        biasv[nt] = gb[e0 + wn * 64 + nt * 16 + laneN];

    float colSum[4] = {0.f, 0.f, 0.f, 0.f};
#pragma unroll
    for (int mt = 0; mt < 2; ++mt) {
#pragma unroll
        for (int r = 0; r < 8; ++r) {
            const int t = t0 + wm * 32 + mt * 16 + r + hi * 8;
            const float aw = attnw[(size_t)b * T + t];
            const float* xrow = x + ((size_t)b * T + t) * (size_t)D + e0 + wn * 64;
#pragma unroll
            for (int nt = 0; nt < 4; ++nt) {
                const float g  = fast_sigmoid(acc[mt][nt][r] + biasv[nt]);
                colSum[nt] += aw * xrow[nt * 16 + laneN] * g;
            }
        }
    }
    // Fold lane halves (same e, different M rows), then fixed-order cross-wave sum.
#pragma unroll
    for (int nt = 0; nt < 4; ++nt) {
        const float cs = colSum[nt] + __shfl_xor(colSum[nt], 16, 32);
        if (hi == 0) red[wm][wn * 64 + nt * 16 + laneN] = cs;
    }
    __syncthreads();
    if (tid < TILE_N) {
        const float s = red[0][tid] + red[1][tid] + red[2][tid] + red[3][tid];
        partial[((size_t)(b * MB + blockIdx.y)) * (size_t)D + e0 + tid] = s;
    }
}

// ---------------------------------------------------------------------------
// Kernel 3: deterministic fixed-order reduction of M-block partials.
// ---------------------------------------------------------------------------
__global__ __launch_bounds__(256)
void gp_reduce_kernel(const float* __restrict__ partial, float* __restrict__ out,
                      int MB, int D, int total) {
    const int i = blockIdx.x * blockDim.x + threadIdx.x;
    if (i >= total) return;
    const int b = i / D;
    const int e = i - b * D;
    float s = 0.f;
    for (int m = 0; m < MB; ++m)
        s += partial[((size_t)(b * MB + m)) * (size_t)D + e];
    out[i] = s;
}

// ---------------------------------------------------------------------------
extern "C" void kernel_launch(void* const* d_in, const int* in_sizes, int n_in,
                              void* d_out, int out_size, void* d_ws, size_t ws_size,
                              hipStream_t stream) {
    const float* x  = (const float*)d_in[0];   // (B, T, D)
    const float* sc = (const float*)d_in[1];   // (B, T, 1)
    const float* gw = (const float*)d_in[2];   // (D, D)
    const float* gb = (const float*)d_in[3];   // (D,)
    float* out = (float*)d_out;                // (B, D)

    const int D  = in_sizes[3];
    const int BT = in_sizes[1];
    const int B  = out_size / D;
    const int T  = BT / B;
    const int MB = T / TILE_M;

    float* attnw   = (float*)d_ws;                 // B*T floats
    float* partial = attnw + (size_t)B * T;        // B*MB*D floats

    gp_entmax_kernel<<<B, ENT_THREADS, 0, stream>>>(sc, attnw, T);

    dim3 grid(D / TILE_N, T / TILE_M, B);
    gp_gemm_pool_kernel<<<grid, 256, 0, stream>>>(x, gw, gb, attnw, partial, T, D, MB);

    const int total = B * D;
    gp_reduce_kernel<<<(total + 255) / 256, 256, 0, stream>>>(partial, out, MB, D, total);
}
